// MobiusEncDecGRUAttn_28819230556834
// MI455X (gfx1250) — matrix operand order repaired and checked by reading, not compiled, not run on hardware
//
#include <hip/hip_runtime.h>
#include <hip/hip_bf16.h>

// ---------------- problem constants ----------------
#define B_      512
#define T_      75
#define IN_     768
#define H_      256
#define H3_     768
#define NSPAN_  8
#define OUT_    2
#define NDAYS_  10
#define TB_     (B_ * T_)      // 38400 rows
#define EPS_    1e-5f
#define MAXN_   (1.0f - 1e-3f)

// ---------------- WMMA types (gfx1250, wave32) ----------------
typedef _Float16 half_t;
typedef half_t v16h __attribute__((ext_vector_type(16)));
typedef float  v8f  __attribute__((ext_vector_type(8)));

// ---------------- block-wide reductions (256 threads) ----------------
__device__ __forceinline__ float block_sum(float v, float* s) {
    int t = threadIdx.x;
    s[t] = v; __syncthreads();
#pragma unroll
    for (int k = 128; k > 0; k >>= 1) { if (t < k) s[t] += s[t + k]; __syncthreads(); }
    float r = s[0]; __syncthreads();
    return r;
}
__device__ __forceinline__ float block_max(float v, float* s) {
    int t = threadIdx.x;
    s[t] = v; __syncthreads();
#pragma unroll
    for (int k = 128; k > 0; k >>= 1) { if (t < k) s[t] = fmaxf(s[t], s[t + k]); __syncthreads(); }
    float r = s[0]; __syncthreads();
    return r;
}

__device__ __forceinline__ float artanh_clip(float x) {
    x = fminf(fmaxf(x, -0.9999999f), 0.9999999f);
    return 0.5f * __logf((1.f + x) / (1.f - x));
}
__device__ __forceinline__ float sigmoidf(float x) { return 1.f / (1.f + __expf(-x)); }

// mobius_add with projection; 4 block reductions
__device__ __forceinline__ float mobius_add_block(float x, float y, float* red) {
    float x2 = block_sum(x * x, red);
    float y2 = block_sum(y * y, red);
    float xy = block_sum(x * y, red);
    float den = fmaxf(1.f + 2.f * xy + x2 * y2, EPS_);
    float v = ((1.f + 2.f * xy + y2) * x + (1.f - x2) * y) / den;
    float n2 = block_sum(v * v, red);
    float n = sqrtf(fmaxf(n2, 1e-15f));
    if (n > MAXN_) v *= MAXN_ / n;
    return v;
}

// ======================================================================
// Generic WMMA GEMM: C[M,N] = act( A[M,K] * W[N,K]^T + bias ), f16 MACs,
// f32 accumulate. A rows addressed as A[abase + row*lda + k]; optional
// compile-time per-row scale (fuses expmap0 projection into staging).
// Tiles: BM=64 BN=32 BK=32; 8 waves; one v_wmma_f32_16x16x32_f16 16x16
// subtile per wave per k-step. LDS double-buffered (one barrier per
// k-step); staging uses 3x global_load_b128 per thread issued together,
// packed ds_store_b64 writes, and prefetch of the next tile.
// Requires M%64==0, N%32==0, K%32==0 (all shapes here comply).
// ======================================================================
#define TM 64
#define TN 32
#define TK 32

template<bool HAS_SROW>
__global__ __launch_bounds__(256) void gemm_f16(
    const float* __restrict__ A, long lda, long abase, const float* __restrict__ srow,
    const float* __restrict__ W, float* __restrict__ C,
    int N, int K, const float* __restrict__ bias, int act)
{
    __shared__ __align__(16) half_t As[2][TM][TK + 8];   // row stride 80B
    __shared__ __align__(16) half_t Bs[2][TN][TK + 8];
    const int tid  = threadIdx.x;
    const int m0   = blockIdx.x * TM;
    const int n0   = blockIdx.y * TN;
    const int ktiles = K / TK;

    // staging map: thread owns a float4 slice; rows rA and rA+32 of A, row rA of B
    const int rA = tid >> 3;          // 0..31
    const int c4 = (tid & 7) * 4;     // 0,4,...,28
    const float* Abase = A + abase + (long)(m0 + rA) * lda + c4;
    const float* Wbase = W + (long)(n0 + rA) * K + c4;
    const float sc0 = HAS_SROW ? srow[m0 + rA]      : 1.f;
    const float sc1 = HAS_SROW ? srow[m0 + rA + 32] : 1.f;

    auto stage = [&](int kt, int buf) {
        const float* Ab = Abase + (long)kt * TK;
        const float* Wb = Wbase + (long)kt * TK;
        // issue all three 128-bit loads before converting
        float4 a0 = *(const float4*)Ab;
        float4 a1 = *(const float4*)(Ab + 32 * lda);
        float4 b0 = *(const float4*)Wb;
        if (HAS_SROW) {
            a0.x *= sc0; a0.y *= sc0; a0.z *= sc0; a0.w *= sc0;
            a1.x *= sc1; a1.y *= sc1; a1.z *= sc1; a1.w *= sc1;
        }
        union { half_t h[4]; uint2 u; } p0, p1, pb;
        p0.h[0] = (half_t)a0.x; p0.h[1] = (half_t)a0.y;
        p0.h[2] = (half_t)a0.z; p0.h[3] = (half_t)a0.w;
        p1.h[0] = (half_t)a1.x; p1.h[1] = (half_t)a1.y;
        p1.h[2] = (half_t)a1.z; p1.h[3] = (half_t)a1.w;
        pb.h[0] = (half_t)b0.x; pb.h[1] = (half_t)b0.y;
        pb.h[2] = (half_t)b0.z; pb.h[3] = (half_t)b0.w;
        *(uint2*)&As[buf][rA][c4]      = p0.u;    // ds_store_b64
        *(uint2*)&As[buf][rA + 32][c4] = p1.u;
        *(uint2*)&Bs[buf][rA][c4]      = pb.u;
        if (kt + 1 < ktiles) {                    // pull tile kt+1 toward WGP
            __builtin_prefetch(Ab + TK, 0, 1);
            __builtin_prefetch(Wb + TK, 0, 1);
        }
    };

    const int wid  = tid >> 5;
    const int lane = tid & 31;
    const int wm   = (wid & 3) * 16;     // wave's row offset in tile
    const int wn   = (wid >> 2) * 16;    // wave's col offset in tile
    const int hi   = (lane < 16) ? 0 : 1;
    const int l16  = lane & 15;

    v8f acc = {};
    stage(0, 0);
    __syncthreads();
    for (int kt = 0; kt < ktiles; ++kt) {
        const int cur = kt & 1;
        if (kt + 1 < ktiles) stage(kt + 1, cur ^ 1);

        // A fragment: lane holds row wm + (lane&15); K pairs per ISA layout:
        // lanes 0-15: V0..3 -> K=0..7, V4..7 -> K=16..23 (contiguous bytes)
        // lanes16-31: V0..3 -> K=8..15, V4..7 -> K=24..31
        v16h afrag, bfrag;
        const int am = wm + l16;
#pragma unroll
        for (int j = 0; j < 8; ++j) {
            int kb = (j < 4) ? (2 * j + (hi ? 8 : 0)) : (16 + 2 * (j - 4) + (hi ? 8 : 0));
            afrag[2 * j + 0] = As[cur][am][kb + 0];
            afrag[2 * j + 1] = As[cur][am][kb + 1];
        }
        // B fragment: lane holds col wn + (lane&15); same K order as A fragment
        const int bn = wn + l16;
        const int koff = hi ? 8 : 0;
#pragma unroll
        for (int e = 0; e < 16; ++e) bfrag[e] = Bs[cur][bn][koff + e + (e >= 8 ? 8 : 0)];

        acc = __builtin_amdgcn_wmma_f32_16x16x32_f16(false, afrag, false, bfrag,
                                                     (short)0, acc, false, false);
        __syncthreads();
    }
    // D layout: VGPR r -> row r (lanes 0-15) / r+8 (lanes 16-31); col = lane&15
#pragma unroll
    for (int r = 0; r < 8; ++r) {
        int row = m0 + wm + r + (hi ? 8 : 0);
        int col = n0 + wn + l16;
        float v = acc[r];
        if (bias) v += bias[col];
        if (act == 1) v = fmaxf(v, 0.f);
        C[(long)row * N + col] = v;
    }
}

// ======================================================================
// Element-wise kernels (one block of 256 threads per row)
// ======================================================================

__global__ __launch_bounds__(256) void fill0(float* p, long n) {
    long i = (long)blockIdx.x * 256 + threadIdx.x;
    if (i < n) p[i] = 0.f;
}

// expmap0 + project fused into a per-row scale; also store ||x_hyp|| clamp.
__global__ __launch_bounds__(256) void rowscale(const float* __restrict__ x,
                                                float* __restrict__ srow,
                                                float* __restrict__ xnh) {
    __shared__ float red[256];
    long r = blockIdx.x;
    float s = 0.f;
    const float* xr = x + r * IN_;
#pragma unroll
    for (int j = 0; j < 3; ++j) { float v = xr[threadIdx.x + 256 * j]; s += v * v; }
    float s2 = block_sum(s, red);
    float nu = sqrtf(fmaxf(s2, 1e-15f));
    float n  = fmaxf(nu, EPS_);
    float f  = tanhf(n) / n;
    float ny = f * nu;
    if (ny > MAXN_) f *= MAXN_ / ny;
    if (threadIdx.x == 0) { srow[r] = f; xnh[r] = fmaxf(f * nu, EPS_); }
}

// finish mobius_matvec(U, x_hyp): per-gate tanh rescale + projection (in place)
__global__ __launch_bounds__(256) void ux_finish(float* __restrict__ ux,
                                                 const float* __restrict__ xnh) {
    __shared__ float red[256];
    long r = blockIdx.x;
    int e = threadIdx.x;
    float xn = xnh[r];
    float at = artanh_clip(xn);
#pragma unroll
    for (int g = 0; g < 3; ++g) {
        long idx = r * H3_ + g * H_ + e;
        float m = ux[idx];
        float m2 = block_sum(m * m, red);
        float mxn = fmaxf(sqrtf(fmaxf(m2, 1e-15f)), EPS_);
        float tv = tanhf(mxn / xn * at);
        float w = tv / mxn * m;
        float nw = fabsf(tv);                 // exact norm after rescale
        if (nw > MAXN_) w *= MAXN_ / nw;
        ux[idx] = w;
    }
}

// hyperbolic bias points: expmap0(enc_bias), 3 rows of H
__global__ __launch_bounds__(256) void bias_prep(const float* __restrict__ eb,
                                                 float* __restrict__ bpts) {
    __shared__ float red[256];
    int e = threadIdx.x;
#pragma unroll
    for (int g = 0; g < 3; ++g) {
        float u = eb[g * H_ + e];
        float s2 = block_sum(u * u, red);
        float nu = sqrtf(fmaxf(s2, 1e-15f));
        float n = fmaxf(nu, EPS_);
        float f = tanhf(n) / n;
        float ny = f * nu;
        if (ny > MAXN_) f *= MAXN_ / ny;
        bpts[g * H_ + e] = f * u;
    }
}

// matvec-rescale -> mobius_add(ux) -> mobius_add(bias) -> logmap0
__device__ __forceinline__ float gate_preact(float m, float u, float bias,
                                             float hn, float* red) {
    float m2 = block_sum(m * m, red);
    float mxn = fmaxf(sqrtf(fmaxf(m2, 1e-15f)), EPS_);
    float tv = tanhf(mxn / hn * artanh_clip(hn));
    float w = tv / mxn * m;
    float nw = fabsf(tv);
    if (nw > MAXN_) w *= MAXN_ / nw;
    float a = mobius_add_block(w, u, red);
    a = mobius_add_block(a, bias, red);
    float n2 = block_sum(a * a, red);
    float n = fmaxf(sqrtf(fmaxf(n2, 1e-15f)), EPS_);
    return artanh_clip(n) / n * a;
}

__global__ __launch_bounds__(256) void enc_gates(
    const float* __restrict__ h_hyp, const float* __restrict__ mmr,
    const float* __restrict__ mmz, const float* __restrict__ ux, int t,
    const float* __restrict__ bpts, float* __restrict__ zbuf, float* __restrict__ rhbuf)
{
    __shared__ float red[256];
    int b = blockIdx.x, e = threadIdx.x;
    long ib = (long)b * H_ + e;
    float h = h_hyp[ib];
    float h2 = block_sum(h * h, red);
    float hn = fmaxf(sqrtf(fmaxf(h2, 1e-15f)), EPS_);
    long ub = ((long)b * T_ + t) * H3_;
    float r = sigmoidf(gate_preact(mmr[ib], ux[ub + e],        bpts[e],        hn, red));
    float z = sigmoidf(gate_preact(mmz[ib], ux[ub + 2*H_ + e], bpts[2*H_ + e], hn, red));
    zbuf[ib] = z;
    // rh = mobius_pointwise_mul(r, h)
    float w = r * h;
    float w2 = block_sum(w * w, red);
    float wxn = fmaxf(sqrtf(fmaxf(w2, 1e-15f)), EPS_);
    float tv = tanhf(wxn / hn * artanh_clip(hn));
    float v = tv / wxn * w;
    float nv = fabsf(tv);
    if (nv > MAXN_) v *= MAXN_ / nv;
    rhbuf[ib] = v;
}

__global__ __launch_bounds__(256) void enc_update(
    float* __restrict__ h_hyp, const float* __restrict__ mmh,
    const float* __restrict__ ux, int t, const float* __restrict__ bpts,
    const float* __restrict__ zbuf, const float* __restrict__ rhbuf,
    float* __restrict__ fullb)
{
    __shared__ float red[256];
    int b = blockIdx.x, e = threadIdx.x;
    long ib = (long)b * H_ + e;
    float h  = h_hyp[ib];
    float rh = rhbuf[ib];
    float z  = zbuf[ib];
    float m  = mmh[ib];
    float rh2 = block_sum(rh * rh, red);
    float rhn = fmaxf(sqrtf(fmaxf(rh2, 1e-15f)), EPS_);
    float m2 = block_sum(m * m, red);
    float mxn = fmaxf(sqrtf(fmaxf(m2, 1e-15f)), EPS_);
    float tv = tanhf(mxn / rhn * artanh_clip(rhn));
    float w = tv / mxn * m;
    float nw = fabsf(tv); if (nw > MAXN_) w *= MAXN_ / nw;
    long ub = ((long)b * T_ + t) * H3_;
    float a  = mobius_add_block(w, ux[ub + H_ + e], red);
    float ht = mobius_add_block(a, bpts[H_ + e], red);
    float d  = mobius_add_block(-h, ht, red);
    // mobius_pointwise_mul(z, delta)
    float d2 = block_sum(d * d, red);
    float dn = fmaxf(sqrtf(fmaxf(d2, 1e-15f)), EPS_);
    float wx = z * d;
    float wx2 = block_sum(wx * wx, red);
    float wxn = fmaxf(sqrtf(fmaxf(wx2, 1e-15f)), EPS_);
    float tv2 = tanhf(wxn / dn * artanh_clip(dn));
    float pw = tv2 / wxn * wx;
    float np = fabsf(tv2); if (np > MAXN_) pw *= MAXN_ / np;
    float hnew = mobius_add_block(h, pw, red);
    h_hyp[ib] = hnew;
    float n2 = block_sum(hnew * hnew, red);
    float n = fmaxf(sqrtf(fmaxf(n2, 1e-15f)), EPS_);
    fullb[((long)b * T_ + t) * H_ + e] = artanh_clip(n) / n * hnew;
}

// attention score per (b,t): v . tanh(p1 + b1 + p2 + b2) + bv
__global__ __launch_bounds__(256) void attn_score(
    const float* __restrict__ p1, const float* __restrict__ p2,
    const float* __restrict__ b1, const float* __restrict__ b2,
    const float* __restrict__ v,  const float* __restrict__ bv,
    float* __restrict__ score)
{
    __shared__ float red[256];
    long bt = blockIdx.x;
    int b = (int)(bt / T_);
    int e = threadIdx.x;
    float s = tanhf(p1[bt * H_ + e] + b1[e] + p2[(long)b * H_ + e] + b2[e]) * v[e];
    float r = block_sum(s, red);
    if (e == 0) score[bt] = r + bv[0];
}

// masked softmax over T + weighted sum of `full` -> ctx
__global__ __launch_bounds__(256) void softmax_ctx(
    const float* __restrict__ score, const int* __restrict__ lens,
    const float* __restrict__ fullb, float* __restrict__ ctx)
{
    __shared__ float red[256];
    __shared__ float wts[T_];
    int b = blockIdx.x, tid = threadIdx.x;
    int len = lens[b];
    float s = -3.0e38f;
    if (tid < T_) s = (tid < len) ? score[(long)b * T_ + tid] : -1e9f;
    float mx = block_max(s, red);
    float e = (tid < T_) ? __expf(s - mx) : 0.f;
    float sum = block_sum(e, red);
    if (tid < T_) wts[tid] = e / sum;
    __syncthreads();
    float acc = 0.f;
    const float* fb = fullb + (long)b * T_ * H_;
    for (int t = 0; t < T_; ++t) acc += wts[t] * fb[t * H_ + tid];
    ctx[(long)b * H_ + tid] = acc;
}

__global__ __launch_bounds__(256) void span_head(
    const float* __restrict__ ctx, const float* __restrict__ sw,
    const float* __restrict__ sb, float* __restrict__ out)
{
    __shared__ float red[256];
    __shared__ float lg[NSPAN_];
    int b = blockIdx.x, e = threadIdx.x;
    float c = ctx[(long)b * H_ + e];
#pragma unroll
    for (int k = 0; k < NSPAN_; ++k) {
        float s = block_sum(c * sw[k * H_ + e], red);
        if (e == 0) lg[k] = s + sb[k];
    }
    __syncthreads();
    if (e == 0) {
        float mx = lg[0];
        for (int k = 1; k < NSPAN_; ++k) mx = fmaxf(mx, lg[k]);
        float ex[NSPAN_], sum = 0.f;
        for (int k = 0; k < NSPAN_; ++k) { ex[k] = __expf(lg[k] - mx); sum += ex[k]; }
        for (int k = 0; k < NSPAN_; ++k) out[(long)b * NSPAN_ + k] = ex[k] / sum;
    }
}

__global__ __launch_bounds__(256) void dec_gate(
    const float* __restrict__ gi, const float* __restrict__ gh,
    const float* __restrict__ bih, const float* __restrict__ bhh,
    float* __restrict__ h)
{
    int b = blockIdx.x, e = threadIdx.x;
    long gb = (long)b * H3_;
    float ir = gi[gb + e]          + bih[e];
    float iz = gi[gb + H_ + e]     + bih[H_ + e];
    float in = gi[gb + 2*H_ + e]   + bih[2*H_ + e];
    float hr = gh[gb + e]          + bhh[e];
    float hz = gh[gb + H_ + e]     + bhh[H_ + e];
    float hn = gh[gb + 2*H_ + e]   + bhh[2*H_ + e];
    long ib = (long)b * H_ + e;
    float hv = h[ib];
    float r = sigmoidf(ir + hr);
    float z = sigmoidf(iz + hz);
    float n = tanhf(in + r * hn);
    h[ib] = (1.f - z) * n + z * hv;
}

__global__ __launch_bounds__(256) void dec_out(
    const float* __restrict__ h, const float* __restrict__ w,
    const float* __restrict__ bo, float* __restrict__ out, int step)
{
    __shared__ float red[256];
    int b = blockIdx.x, e = threadIdx.x;
    float hv = h[(long)b * H_ + e];
    float s0 = block_sum(hv * w[e], red);
    float s1 = block_sum(hv * w[H_ + e], red);
    if (e == 0) {
        float o0 = s0 + bo[0], o1 = s1 + bo[1];
        float mx = fmaxf(o0, o1);
        float e0 = __expf(o0 - mx), e1 = __expf(o1 - mx);
        float inv = 1.f / (e0 + e1);
        long base = (long)B_ * NSPAN_ + (long)b * NDAYS_ * OUT_ + (long)step * OUT_;
        out[base] = e0 * inv;
        out[base + 1] = e1 * inv;
    }
}

// ---------------- host side ----------------
static inline void launch_gemm(hipStream_t s, const float* A, long lda, long abase,
                               const float* srow, const float* W, float* C,
                               int M, int N, int K, const float* bias, int act) {
    dim3 grid(M / TM, N / TN);
    if (srow)
        gemm_f16<true><<<grid, 256, 0, s>>>(A, lda, abase, srow, W, C, N, K, bias, act);
    else
        gemm_f16<false><<<grid, 256, 0, s>>>(A, lda, abase, nullptr, W, C, N, K, bias, act);
}

extern "C" void kernel_launch(void* const* d_in, const int* in_sizes, int n_in,
                              void* d_out, int out_size, void* d_ws, size_t ws_size,
                              hipStream_t stream) {
    const float* x        = (const float*)d_in[0];
    const int*   lens     = (const int*)  d_in[1];
    const float* enc_w_ih = (const float*)d_in[2];
    const float* enc_w_hh = (const float*)d_in[3];
    const float* enc_bias = (const float*)d_in[4];
    const float* dec_w_ih = (const float*)d_in[5];
    const float* dec_w_hh = (const float*)d_in[6];
    const float* dec_b_ih = (const float*)d_in[7];
    const float* dec_b_hh = (const float*)d_in[8];
    const float* attn_w1  = (const float*)d_in[9];
    const float* attn_b1  = (const float*)d_in[10];
    const float* attn_w2  = (const float*)d_in[11];
    const float* attn_b2  = (const float*)d_in[12];
    const float* attn_v   = (const float*)d_in[13];
    const float* attn_bv  = (const float*)d_in[14];
    const float* fc_in_w  = (const float*)d_in[15];
    const float* fc_in_b  = (const float*)d_in[16];
    const float* fc_out_w = (const float*)d_in[17];
    const float* fc_out_b = (const float*)d_in[18];
    const float* span_w   = (const float*)d_in[19];
    const float* span_b   = (const float*)d_in[20];
    float* out = (float*)d_out;

    // workspace bump allocator (f32 elements)
    float* ws = (float*)d_ws;
    long off = 0;
    auto alloc = [&](long n) { float* p = ws + off; off += n; return p; };
    float* srow  = alloc(TB_);
    float* xnh   = alloc(TB_);
    float* bpts  = alloc(3L * H_);
    float* h_hyp = alloc((long)B_ * H_);
    float* mmr   = alloc((long)B_ * H_);
    float* mmz   = alloc((long)B_ * H_);
    float* mmh   = alloc((long)B_ * H_);
    float* zbuf  = alloc((long)B_ * H_);
    float* rhbuf = alloc((long)B_ * H_);
    float* fullb = alloc((long)B_ * T_ * H_);
    float* proj2 = alloc((long)B_ * H_);
    float* score = alloc((long)B_ * T_);
    float* ctx   = alloc((long)B_ * H_);        // becomes decoder h
    float* gi    = alloc((long)B_ * H3_);
    float* gh    = alloc((long)B_ * H3_);
    float* inp   = alloc((long)B_ * IN_);
    float* ux    = alloc((long)TB_ * H3_);      // big; dead after encoder
    float* proj1 = ux;                          // alias: attention proj reuses it

    // ---- encoder prep ----
    rowscale<<<TB_, 256, 0, stream>>>(x, srow, xnh);
    bias_prep<<<1, 256, 0, stream>>>(enc_bias, bpts);
    {
        long n = (long)B_ * H_;
        fill0<<<(int)((n + 255) / 256), 256, 0, stream>>>(h_hyp, n);
    }
    // Ux = mobius_matvec(enc_w_ih, expmap0(x)) for all (b,t): one 38400x768x768 GEMM
    launch_gemm(stream, x, IN_, 0, srow, enc_w_ih, ux, TB_, H3_, IN_, nullptr, 0);
    ux_finish<<<TB_, 256, 0, stream>>>(ux, xnh);

    // ---- Mobius GRU scan (sequential over T) ----
    for (int t = 0; t < T_; ++t) {
        // r-gate rows [0:256), z-gate rows [512:768) of enc_w_hh
        launch_gemm(stream, h_hyp, H_, 0, nullptr, enc_w_hh,                 mmr, B_, H_, H_, nullptr, 0);
        launch_gemm(stream, h_hyp, H_, 0, nullptr, enc_w_hh + (long)2*H_*H_, mmz, B_, H_, H_, nullptr, 0);
        enc_gates<<<B_, 256, 0, stream>>>(h_hyp, mmr, mmz, ux, t, bpts, zbuf, rhbuf);
        // candidate rows [256:512)
        launch_gemm(stream, rhbuf, H_, 0, nullptr, enc_w_hh + (long)H_*H_,   mmh, B_, H_, H_, nullptr, 0);
        enc_update<<<B_, 256, 0, stream>>>(h_hyp, mmh, ux, t, bpts, zbuf, rhbuf, fullb);
    }

    // ---- attention ----
    launch_gemm(stream, fullb, H_, 0, nullptr, attn_w1, proj1, TB_, H_, H_, nullptr, 0);
    // hx = logmap0(h_last) == fullb[:, T-1, :] (strided rows)
    launch_gemm(stream, fullb, (long)T_ * H_, (long)(T_ - 1) * H_, nullptr,
                attn_w2, proj2, B_, H_, H_, nullptr, 0);
    attn_score<<<TB_, 256, 0, stream>>>(proj1, proj2, attn_b1, attn_b2, attn_v, attn_bv, score);
    softmax_ctx<<<B_, 256, 0, stream>>>(score, lens, fullb, ctx);
    span_head<<<B_, 256, 0, stream>>>(ctx, span_w, span_b, out);

    // ---- decoder (10 sequential GRUCell steps), h lives in `ctx` ----
    {
        long n = (long)B_ * IN_;
        fill0<<<(int)((n + 255) / 256), 256, 0, stream>>>(inp, n);
    }
    for (int s = 0; s < NDAYS_; ++s) {
        launch_gemm(stream, inp, IN_, 0, nullptr, dec_w_ih, gi, B_, H3_, IN_, nullptr, 0);
        launch_gemm(stream, ctx, H_,  0, nullptr, dec_w_hh, gh, B_, H3_, H_,  nullptr, 0);
        dec_gate<<<B_, 256, 0, stream>>>(gi, gh, dec_b_ih, dec_b_hh, ctx);
        dec_out<<<B_, 256, 0, stream>>>(ctx, fc_out_w, fc_out_b, out, s);
        launch_gemm(stream, ctx, H_, 0, nullptr, fc_in_w, inp, B_, IN_, H_, fc_in_b, 1);
    }
}
